// MoNet_75694503625293
// MI455X (gfx1250) — matrix operand order repaired and checked by reading, not compiled, not run on hardware
//
#include <hip/hip_runtime.h>
#include <hip/hip_bf16.h>

// ---------------------------------------------------------------------------
// MoNet (GMMConv x3 + MLP head) for MI455X / gfx1250, wave32 + WMMA bf16.
// All GEMM operands padded/pre-swizzled so the WMMA loop has zero branches
// and only 16B vector loads.
// ---------------------------------------------------------------------------

typedef __attribute__((ext_vector_type(16))) __bf16 v16bf;
typedef __attribute__((ext_vector_type(8)))  __bf16 v8bf;
typedef __attribute__((ext_vector_type(8)))  float  v8f;

#define KGM 25   // Gaussian mixture components
#define TB  256
#define WT  4    // 16-wide column tiles per wave (wave computes 16x64)

// ------------------------------- utility -----------------------------------

__global__ void zero_f32(float* __restrict__ p, long long n) {
    long long i = (long long)blockIdx.x * blockDim.x + threadIdx.x;
    if (i < n) p[i] = 0.0f;
}

// Weight prep: in f32 [Kd, Nc] row-major  ->  out bf16 [Npad][Kpad] with
// out[col*Kpad + k] = in[k*Nc + col], zero-filled for k>=Kd or col>=Nc.
// Gives each WMMA lane a contiguous, aligned K-run for its B fragment and
// makes K/N padding correct by construction.
__global__ void cvt_transpose_bf16(const float* __restrict__ in, __bf16* __restrict__ out,
                                   int Kd, int Nc, int Kpad, long long total) {
    long long t = (long long)blockIdx.x * blockDim.x + threadIdx.x;
    if (t >= total) return;
    int k = (int)(t % Kpad);
    long long col = t / Kpad;
    float v = 0.0f;
    if (k < Kd && col < Nc) v = in[(long long)k * Nc + col];
    out[t] = (__bf16)v;
}

__global__ void degree_kernel(const long long* __restrict__ dst, float* __restrict__ deg, int E) {
    int e = blockIdx.x * blockDim.x + threadIdx.x;
    if (e < E) atomicAdd(&deg[(int)dst[e]], 1.0f);
}

// ------------------------------- fc0 + ELU ---------------------------------
__global__ void fc0_kernel(const float* __restrict__ x, const float* __restrict__ w,
                           const float* __restrict__ b, __bf16* __restrict__ Hb, int n) {
    long long t = (long long)blockIdx.x * blockDim.x + threadIdx.x;
    if (t >= (long long)n * 16) return;
    int j = (int)(t & 15);
    int i = (int)(t >> 4);
    float v = b[j];
    v = fmaf(x[i * 3 + 0], w[0 * 16 + j], v);
    v = fmaf(x[i * 3 + 1], w[1 * 16 + j], v);
    v = fmaf(x[i * 3 + 2], w[2 * 16 + j], v);
    v = (v > 0.0f) ? v : expm1f(v);
    Hb[(long long)i * 16 + j] = (__bf16)v;
}

// ------------------------- Gaussian edge weights ---------------------------
__global__ void gauss_kernel(const float* __restrict__ ea, const float* __restrict__ mu,
                             const float* __restrict__ sg, float* __restrict__ W, int E) {
    int e = blockIdx.x * blockDim.x + threadIdx.x;
    if (e >= E) return;
    float a0 = ea[(long long)e * 3 + 0];
    float a1 = ea[(long long)e * 3 + 1];
    float a2 = ea[(long long)e * 3 + 2];
#pragma unroll
    for (int k = 0; k < KGM; ++k) {
        float s = 0.0f;
        float d0 = a0 - mu[k * 3 + 0], g0 = sg[k * 3 + 0];
        float d1 = a1 - mu[k * 3 + 1], g1 = sg[k * 3 + 1];
        float d2 = a2 - mu[k * 3 + 2], g2 = sg[k * 3 + 2];
        s -= 0.5f * d0 * d0 / (1e-14f + g0 * g0);
        s -= 0.5f * d1 * d1 / (1e-14f + g1 * g1);
        s -= 0.5f * d2 * d2 / (1e-14f + g2 * g2);
        W[(long long)e * KGM + k] = __expf(s);
    }
}

// ----------------------------- WMMA GEMM -----------------------------------
// C[M, Ncols] = A[M, Kpad-ish] (bf16 row-major, lda) @ B (pre-transposed bf16
// Bt[Ncols][Kpad], zero-padded). M % 16 == 0, Ncols % 64 == 0 (padded by us),
// Kpad % 32 == 0. One wave -> 16x64 tile (4 accumulators, A reused 4x).
// Fragment layouts per CDNA5 ISA 7.12.2 (wave32):
//   A: g=lane>>4, m=lane&15; elems 0..7 -> K=kk+8g+e, elems 8..15 -> K=kk+16+8g+(e-8)
//   B: n=lane&15, same K order as A (two 8-elem runs from Bt)
//   C/D: elem v -> row row0+8g+v, col lane&15
__global__ __launch_bounds__(TB) void gemm_bf16_wmma(
    const __bf16* __restrict__ A, int lda,
    const __bf16* __restrict__ Bt, int Kpad,
    float* __restrict__ Cf, __bf16* __restrict__ Cb, int ldc,
    int M, int Ncols) {
    const int lane = threadIdx.x & 31;
    const int wave = threadIdx.x >> 5;
    const int col0 = (blockIdx.x * 8 + wave) * (16 * WT);
    const int row0 = blockIdx.y * 16;
    if (col0 >= Ncols) return;                    // wave-uniform early out

    const int g = lane >> 4;
    const int m = lane & 15;
    const __bf16* ap  = A  + (size_t)(row0 + m) * lda  + 8 * g;
    const __bf16* bp0 = Bt + (size_t)(col0 + m) * Kpad + 8 * g;

    v8f c[WT] = {};
    for (int kk = 0; kk < Kpad; kk += 32) {
        v8bf alo = *(const v8bf*)ap;
        v8bf ahi = *(const v8bf*)(ap + 16);
        v16bf a = __builtin_shufflevector(alo, ahi, 0, 1, 2, 3, 4, 5, 6, 7,
                                          8, 9, 10, 11, 12, 13, 14, 15);
#pragma unroll
        for (int w = 0; w < WT; ++w) {
            const __bf16* bp = bp0 + (size_t)(16 * w) * Kpad + kk;
            v8bf blo = *(const v8bf*)bp;
            v8bf bhi = *(const v8bf*)(bp + 16);
            v16bf b = __builtin_shufflevector(blo, bhi, 0, 1, 2, 3, 4, 5, 6, 7,
                                              8, 9, 10, 11, 12, 13, 14, 15);
            c[w] = __builtin_amdgcn_wmma_f32_16x16x32_bf16(false, a, false, b,
                                                           (short)0, c[w], false, false);
        }
        ap += 32;
    }

    const int crow = row0 + 8 * g;
#pragma unroll
    for (int w = 0; w < WT; ++w) {
        const int ccol = col0 + 16 * w + m;
#pragma unroll
        for (int v = 0; v < 8; ++v) {
            size_t idx = (size_t)(crow + v) * ldc + ccol;
            if (Cf) Cf[idx] = c[w][v];
            if (Cb) Cb[idx] = (__bf16)c[w][v];
        }
    }
}

// --------------------------- edge messages ---------------------------------
// thread t -> (e, co): Agg[dst[e],co] += sum_k W[e,k] * Xl[src[e], k*Co + co]
__global__ __launch_bounds__(TB) void message_kernel(
    const long long* __restrict__ src, const long long* __restrict__ dst,
    const float* __restrict__ W, const __bf16* __restrict__ Xl, int ldXl,
    float* __restrict__ Agg, int E, int shift) {
    long long t = (long long)blockIdx.x * blockDim.x + threadIdx.x;
    long long total = (long long)E << shift;
    if (t >= total) return;
    int co = (int)(t & ((1 << shift) - 1));
    int e  = (int)(t >> shift);
    int s  = (int)src[e];
    int d  = (int)dst[e];
    const float*  w  = W + (long long)e * KGM;
    const __bf16* xs = Xl + (long long)s * ldXl + co;
    float acc = 0.0f;
#pragma unroll
    for (int k = 0; k < KGM; ++k)
        acc = fmaf(w[k], (float)xs[(long long)k << shift], acc);
    atomicAdd(&Agg[((long long)d << shift) + co], acc);
}

// ----------------------------- finalize ------------------------------------
__global__ void finalize_kernel(const float* __restrict__ Agg, const float* __restrict__ Xr,
                                int ldXr, const float* __restrict__ deg,
                                const float* __restrict__ bias,
                                __bf16* __restrict__ Hb, int n, int Co) {
    long long t = (long long)blockIdx.x * blockDim.x + threadIdx.x;
    if (t >= (long long)n * Co) return;
    int co = (int)(t % Co);
    int i  = (int)(t / Co);
    float dg = deg[i];
    dg = dg < 1.0f ? 1.0f : dg;
    float v = Agg[t] / dg + Xr[(long long)i * ldXr + co] + bias[co];
    v = (v > 0.0f) ? v : expm1f(v);
    Hb[t] = (__bf16)v;
}

__global__ void bias_elu_kernel(const float* __restrict__ F, const float* __restrict__ bias,
                                __bf16* __restrict__ Fb, int n, int Co) {
    long long t = (long long)blockIdx.x * blockDim.x + threadIdx.x;
    if (t >= (long long)n * Co) return;
    int co = (int)(t % Co);
    float v = F[t] + bias[co];
    v = (v > 0.0f) ? v : expm1f(v);
    Fb[t] = (__bf16)v;
}

__global__ void logsoftmax_kernel(const float* __restrict__ L, int ldL,
                                  const float* __restrict__ bias,
                                  float* __restrict__ out, int n) {
    int i = blockIdx.x * blockDim.x + threadIdx.x;
    if (i >= n) return;
    float v[10];
    float mx = -3.4e38f;
#pragma unroll
    for (int j = 0; j < 10; ++j) { v[j] = L[(long long)i * ldL + j] + bias[j]; mx = fmaxf(mx, v[j]); }
    float s = 0.0f;
#pragma unroll
    for (int j = 0; j < 10; ++j) s += __expf(v[j] - mx);
    float ls = __logf(s);
#pragma unroll
    for (int j = 0; j < 10; ++j) out[(long long)i * 10 + j] = v[j] - mx - ls;
}

// ------------------------------- driver ------------------------------------

static inline int cdiv_ll(long long a, long long b) { return (int)((a + b - 1) / b); }
static inline int rnd(int x, int a) { return (x + a - 1) & ~(a - 1); }

extern "C" void kernel_launch(void* const* d_in, const int* in_sizes, int n_in,
                              void* d_out, int out_size, void* d_ws, size_t ws_size,
                              hipStream_t stream) {
    const float*     x     = (const float*)d_in[0];
    const long long* ei    = (const long long*)d_in[1];
    const float*     ea    = (const float*)d_in[2];
    const float*     fc0_w = (const float*)d_in[3];
    const float*     fc0_b = (const float*)d_in[4];
    const float*     fc1_w = (const float*)d_in[5];
    const float*     fc1_b = (const float*)d_in[6];
    const float*     fc2_w = (const float*)d_in[7];
    const float*     fc2_b = (const float*)d_in[8];

    const int N = in_sizes[0] / 3;   // 50000, multiple of 16
    const int E = in_sizes[1] / 2;
    const long long* src = ei;
    const long long* dst = ei + E;

    // workspace layout (256B aligned slots, sized for the widest layer)
    char*  ws  = (char*)d_ws;
    size_t off = 0;
    auto alloc = [&](size_t bytes) -> void* {
        void* p = ws + off;
        off = (off + bytes + 255) & ~(size_t)255;
        return p;
    };
    float*  Wg  = (float*) alloc((size_t)E * KGM * 4);           // gaussian weights [E,K]
    float*  deg = (float*) alloc((size_t)N * 4);                  // in-degree
    __bf16* Hb  = (__bf16*)alloc((size_t)N * 128 * 2 + 256);      // node features bf16 (+K over-read pad)
    __bf16* Xl  = (__bf16*)alloc((size_t)N * 3200 * 2);           // x_l bf16, padded ld
    float*  Xr  = (float*) alloc((size_t)N * 128 * 4);            // x @ root, padded ld
    float*  Agg = (float*) alloc((size_t)N * 128 * 4);            // edge aggregate
    __bf16* Bt  = (__bf16*)alloc((size_t)3200 * 64 * 2 + (size_t)256 * 256 * 2); // transposed weights
    float*  F1  = (float*) alloc((size_t)N * 256 * 4);            // fc1 pre-activation
    __bf16* F1b = (__bf16*)alloc((size_t)N * 256 * 2 + 256);      // fc1 output bf16
    float*  LG  = (float*) alloc((size_t)N * 64 * 4);             // fc2 logits (ld 64)
    (void)ws_size; (void)n_in; (void)out_size;

    auto prepB = [&](const float* w, int Kd, int Nc, int Kpad, int Npad) {
        long long tot = (long long)Npad * Kpad;
        cvt_transpose_bf16<<<cdiv_ll(tot, TB), TB, 0, stream>>>(w, Bt, Kd, Nc, Kpad, tot);
    };
    auto gemm = [&](const __bf16* A, int lda, int Kpad,
                    float* Cf, __bf16* Cb, int ldc, int Ncols) {
        dim3 grid((unsigned)cdiv_ll(Ncols, 16 * WT * 8), (unsigned)(N / 16));
        gemm_bf16_wmma<<<grid, TB, 0, stream>>>(A, lda, Bt, Kpad, Cf, Cb, ldc, N, Ncols);
    };

    // degree (shared by all three conv layers)
    zero_f32<<<cdiv_ll(N, TB), TB, 0, stream>>>(deg, N);
    degree_kernel<<<cdiv_ll(E, TB), TB, 0, stream>>>(dst, deg, E);

    // fc0 + ELU -> Hb [N,16]
    fc0_kernel<<<cdiv_ll((long long)N * 16, TB), TB, 0, stream>>>(x, fc0_w, fc0_b, Hb, N);

    int Ci = 16;
    for (int layer = 0; layer < 3; ++layer) {
        int Co    = 32 << layer;
        int shift = 5 + layer;
        int KC    = KGM * Co;
        int Kpad  = rnd(Ci, 32);
        int KCp   = rnd(KC, 64);      // 832 / 1600 / 3200
        int Cop   = rnd(Co, 64);      // 64 / 64 / 128
        int base  = 9 + 5 * layer;
        const float* mu = (const float*)d_in[base + 0];
        const float* sg = (const float*)d_in[base + 1];
        const float* gw = (const float*)d_in[base + 2];
        const float* rt = (const float*)d_in[base + 3];
        const float* bi = (const float*)d_in[base + 4];

        // Xl = H @ g  (WMMA, bf16 out to halve edge-gather traffic)
        prepB(gw, Ci, KC, Kpad, KCp);
        gemm(Hb, Ci, Kpad, nullptr, Xl, KCp, KCp);
        // Xr = H @ root (WMMA, f32 out)
        prepB(rt, Ci, Co, Kpad, Cop);
        gemm(Hb, Ci, Kpad, Xr, nullptr, Cop, Cop);

        // gaussian weights + edge aggregation
        gauss_kernel<<<cdiv_ll(E, TB), TB, 0, stream>>>(ea, mu, sg, Wg, E);
        zero_f32<<<cdiv_ll((long long)N * Co, TB), TB, 0, stream>>>(Agg, (long long)N * Co);
        message_kernel<<<cdiv_ll((long long)E * Co, TB), TB, 0, stream>>>(
            src, dst, Wg, Xl, KCp, Agg, E, shift);

        // mean + root + bias + ELU -> next Hb
        finalize_kernel<<<cdiv_ll((long long)N * Co, TB), TB, 0, stream>>>(
            Agg, Xr, Cop, deg, bi, Hb, N, Co);
        Ci = Co;
    }

    // fc1 + ELU
    prepB(fc1_w, 128, 256, 128, 256);
    gemm(Hb, 128, 128, F1, nullptr, 256, 256);
    bias_elu_kernel<<<cdiv_ll((long long)N * 256, TB), TB, 0, stream>>>(F1, fc1_b, F1b, N, 256);

    // fc2 + log_softmax
    prepB(fc2_w, 256, 10, 256, 64);
    gemm(F1b, 256, 256, LG, nullptr, 64, 64);
    logsoftmax_kernel<<<cdiv_ll(N, TB), TB, 0, stream>>>(LG, 64, fc2_b, (float*)d_out, N);
}
